// DynamicGNN_13365938226087
// MI455X (gfx1250) — compile-verified, matrix operand order checked by reading, not run on hardware
//
#include <hip/hip_runtime.h>
#include <hip/hip_bf16.h>
#include <cstdint>

// ---------------------------------------------------------------------------
// Problem dimensions (fixed by the reference)
// ---------------------------------------------------------------------------
#define BB 4
#define SS 30
#define NN 2000
#define DD 32
#define HD 64
#define HEADS 4
#define EE 128000
#define BN (BB * NN)   // 8000 sequences / graph nodes
#define GG (4 * HD)    // 256 LSTM gate width
#define HO (2 * HD)    // 128 bidirectional hidden width

typedef __attribute__((ext_vector_type(16))) _Float16 v16h;
typedef __attribute__((ext_vector_type(8)))  float    v8f;

// Wave-local LDS fence: per-wave staging buffers are only read by the same
// wave, so waiting on our own DS stores is sufficient (split DScnt counter).
__device__ __forceinline__ void wave_lds_fence() {
    asm volatile("s_wait_dscnt 0" ::: "memory");
}

// Fast transcendentals for the LSTM recurrence (serial critical path):
// v_rcp_f32 instead of IEEE divide; tanh via exp2-based __expf + rcp.
__device__ __forceinline__ float fast_rcp(float x) { return __builtin_amdgcn_rcpf(x); }
__device__ __forceinline__ float sigmf(float x) { return fast_rcp(1.f + __expf(-x)); }
__device__ __forceinline__ float tanh_fast(float x) {
    return 1.f - 2.f * fast_rcp(__expf(2.f * x) + 1.f);
}

// ---------------------------------------------------------------------------
// WMMA helpers: D(16x16 f32) += A(16xK f16) * W(16xK f16)^T
// Both operands are row-major 16xK tiles in LDS.
// 16-bit A-fragment layout (CDNA5 ISA 7.12.2):
//   lane l: row = l&15;  l<16 -> K in {0..7, 16..23}; l>=16 -> K in {8..15, 24..31}
// B uses the mirrored layout (lane -> column of W-tile row).
// ---------------------------------------------------------------------------
__device__ __forceinline__ v16h load_frag16(const _Float16* base, int rowStride,
                                            int k0, int lane) {
    const int r  = lane & 15;
    const int hb = (lane >> 4) << 3;  // 0 or 8
    const _Float16* p = base + r * rowStride + k0;
    v16h v;
#pragma unroll
    for (int e = 0; e < 8; ++e) v[e] = p[hb + e];
#pragma unroll
    for (int e = 0; e < 8; ++e) v[8 + e] = p[16 + hb + e];
    return v;
}

template <int K>
__device__ __forceinline__ v8f wave_gemm16(const _Float16* A, int aStride,
                                           const _Float16* W, int wStride,
                                           v8f acc, int lane) {
#pragma unroll
    for (int k0 = 0; k0 < K; k0 += 32) {
        v16h a = load_frag16(A, aStride, k0, lane);
        v16h b = load_frag16(W, wStride, k0, lane);
        acc = __builtin_amdgcn_wmma_f32_16x16x32_f16(
            false, a, false, b, (short)0, acc, false, false);
    }
    return acc;
}

// ---------------------------------------------------------------------------
// Fused bidirectional-LSTM direction kernel.
// One block owns 16 sequences for all 30 timesteps. Per step:
//   gates(16x256) = [x_t | h](16x(DIN+64)) @ Wcat(256x(DIN+64))^T + bias
// Wcat, h, c live in LDS for the whole kernel.
// ---------------------------------------------------------------------------
template <int DIN, bool GATHER_X>
__global__ void lstm_seq_kernel(const float* __restrict__ xseq,      // (B,S,N,D) if GATHER_X
                                const _Float16* __restrict__ hin,    // (BN,S,HO) if !GATHER_X
                                const float* __restrict__ W_ih,      // (GG, DIN)
                                const float* __restrict__ W_hh,      // (GG, HD)
                                const float* __restrict__ bias,      // (GG)
                                _Float16* __restrict__ hout,         // (BN,S,HO)
                                int dirOff, int reverse) {
    constexpr int K = DIN + HD;
    extern __shared__ char smem[];
    _Float16* Wcat  = (_Float16*)smem;                                   // GG*K
    _Float16* Atile = (_Float16*)(smem + (size_t)GG * K * 2);            // 16*K (cols DIN.. = h)
    float*    gates = (float*)(smem + (size_t)GG * K * 2 + 16 * K * 2);  // 16*GG
    float*    cst   = gates + 16 * GG;                                   // 16*HD
    float*    bl    = cst + 16 * HD;                                     // GG

    const int tid  = threadIdx.x;
    const int lane = tid & 31;
    const int wave = tid >> 5;   // blockDim.x == 256 -> 8 waves
    const int r0   = blockIdx.x * 16;

    // Stage concatenated weights [W_ih | W_hh] as f16 (once).
    for (int idx = tid; idx < GG * K; idx += blockDim.x) {
        int g = idx / K, k = idx % K;
        float v = (k < DIN) ? W_ih[g * DIN + k] : W_hh[g * HD + (k - DIN)];
        Wcat[idx] = (_Float16)v;
    }
    for (int idx = tid; idx < GG; idx += blockDim.x) bl[idx] = bias[idx];
    for (int idx = tid; idx < 16 * HD; idx += blockDim.x) {
        int m = idx / HD, j = idx % HD;
        cst[idx] = 0.f;
        Atile[m * K + DIN + j] = (_Float16)0.f;
    }
    __syncthreads();

    for (int step = 0; step < SS; ++step) {
        const int t = reverse ? (SS - 1 - step) : step;

        // Stage x_t into A cols [0, DIN).
        for (int idx = tid; idx < 16 * DIN; idx += blockDim.x) {
            int m = idx / DIN, k = idx % DIN;
            int row = r0 + m;
            float v;
            if (GATHER_X) {
                int b = row / NN, n = row % NN;
                v = xseq[(((size_t)b * SS + t) * NN + n) * DD + k];
            } else {
                v = (float)hin[(size_t)row * SS * HO + (size_t)t * HO + k];
            }
            Atile[m * K + k] = (_Float16)v;
        }
        // Prefetch next timestep's input rows (global_prefetch_b8) so the
        // gather latency hides behind this step's GEMM + cell math.
        if (step + 1 < SS) {
            const int tn = reverse ? (SS - 2 - step) : (step + 1);
            if (tid < 16) {
                int row = r0 + tid;
                const void* p;
                if (GATHER_X) {
                    int b = row / NN, n = row % NN;
                    p = &xseq[(((size_t)b * SS + tn) * NN + n) * DD];
                } else {
                    p = &hin[(size_t)row * SS * HO + (size_t)tn * HO];
                }
                __builtin_prefetch(p, 0, 0);
            }
        }
        __syncthreads();

        // 16 N-tiles over the 256 gate columns: wave w does tiles w and w+8.
#pragma unroll
        for (int sub = 0; sub < 2; ++sub) {
            const int nt = wave + sub * 8;
            v8f acc;
            float bv = bl[nt * 16 + (lane & 15)];
#pragma unroll
            for (int v = 0; v < 8; ++v) acc[v] = bv;
            acc = wave_gemm16<K>(Atile, K, Wcat + nt * 16 * K, K, acc, lane);
#pragma unroll
            for (int v = 0; v < 8; ++v) {
                int m = ((lane >> 4) << 3) + v;
                int n = lane & 15;
                gates[m * GG + nt * 16 + n] = acc[v];
            }
        }
        __syncthreads();

        // LSTM cell elementwise; update h (in Atile) and c; emit h to global.
        for (int idx = tid; idx < 16 * HD; idx += blockDim.x) {
            int m = idx / HD, j = idx % HD;
            float gi = gates[m * GG + j];
            float gf = gates[m * GG + HD + j];
            float gg = gates[m * GG + 2 * HD + j];
            float go = gates[m * GG + 3 * HD + j];
            float c  = sigmf(gf) * cst[idx] + sigmf(gi) * tanh_fast(gg);
            float h  = sigmf(go) * tanh_fast(c);
            cst[idx] = c;
            Atile[m * K + DIN + j] = (_Float16)h;
            hout[(size_t)(r0 + m) * SS * HO + (size_t)t * HO + dirOff + j] = (_Float16)h;
        }
        __syncthreads();
    }
}

// ---------------------------------------------------------------------------
// GAT projection GEMM: out(BN x 256 f32) = A(BN x K f16) @ gW(K x 256 f32)
// A tile is copied memory->LDS with async-to-LDS b128 transfers (ASYNCcnt);
// gW is staged transposed per N-tile into per-wave LDS buffers.
// ---------------------------------------------------------------------------
template <int K>
__global__ void gemm_a16_wt_kernel(const _Float16* __restrict__ Asrc,
                                   long rowStride,               // in halves
                                   const float* __restrict__ W,  // (K, GG) row-major
                                   float* __restrict__ out) {    // (BN, GG)
    __shared__ alignas(16) _Float16 Atile[16 * K];
    __shared__ alignas(16) _Float16 Wt[8][16 * K];
    const int tid = threadIdx.x, lane = tid & 31, wave = tid >> 5;
    const int r0 = blockIdx.x * 16;

    // Async copy A tile: rows are contiguous f16, 16B aligned -> b128 chunks.
    {
        constexpr int CPR    = (K * 2) / 16;  // 16B chunks per row
        constexpr int CHUNKS = 16 * CPR;      // total chunks (<= 256)
        const int c = tid;
        if (c < CHUNKS) {
            const int m = c / CPR, kc = c % CPR;
            const char* g = (const char*)(Asrc + (size_t)(r0 + m) * rowStride) + kc * 16;
            unsigned lds = (unsigned)(uintptr_t)((char*)Atile + c * 16);
            asm volatile("global_load_async_to_lds_b128 %0, %1, off"
                         :: "v"(lds), "v"(g) : "memory");
        }
        asm volatile("s_wait_asynccnt 0" ::: "memory");
    }
    __syncthreads();

#pragma unroll
    for (int sub = 0; sub < 2; ++sub) {
        const int nt = wave + sub * 8;
        for (int idx = lane; idx < 16 * K; idx += 32) {
            int n = idx / K, k = idx % K;
            Wt[wave][idx] = (_Float16)W[(size_t)k * GG + nt * 16 + n];  // transpose stage
        }
        wave_lds_fence();
        v8f acc;
#pragma unroll
        for (int v = 0; v < 8; ++v) acc[v] = 0.f;
        acc = wave_gemm16<K>(Atile, K, Wt[wave], K, acc, lane);
#pragma unroll
        for (int v = 0; v < 8; ++v) {
            int m = ((lane >> 4) << 3) + v;
            int n = lane & 15;
            out[(size_t)(r0 + m) * GG + nt * 16 + n] = acc[v];
        }
    }
}

// ---------------------------------------------------------------------------
// GAT attention pieces
// ---------------------------------------------------------------------------
__global__ void alpha_kernel(const float* __restrict__ xp,
                             const float* __restrict__ ga_src,
                             const float* __restrict__ ga_dst,
                             float* __restrict__ asrc, float* __restrict__ adst) {
    int i = blockIdx.x * blockDim.x + threadIdx.x;
    if (i >= BN * HEADS) return;
    int n = i / HEADS, h = i % HEADS;
    const float* xph = xp + (size_t)n * GG + h * HD;
    float s = 0.f, d = 0.f;
    for (int c = 0; c < HD; ++c) {
        s += xph[c] * ga_src[h * HD + c];
        d += xph[c] * ga_dst[h * HD + c];
    }
    asrc[i] = s;
    adst[i] = d;
}

__device__ __forceinline__ void edge_sd(int ge, const int* __restrict__ ei,
                                        int& s, int& d) {
    const int NE = BB * EE;
    if (ge < NE) {
        int b = ge / EE, e = ge % EE;
        s = ei[e] + b * NN;
        d = ei[EE + e] + b * NN;
    } else {
        s = d = ge - NE;   // self loops
    }
}

__device__ __forceinline__ float leaky02(float x) { return x >= 0.f ? x : 0.2f * x; }

__global__ void edge_max_kernel(const int* __restrict__ ei,
                                const float* __restrict__ asrc,
                                const float* __restrict__ adst,
                                unsigned* __restrict__ emax) {
    const int TOT = (BB * EE + BN) * HEADS;
    int i = blockIdx.x * blockDim.x + threadIdx.x;
    if (i >= TOT) return;
    int ge = i / HEADS, h = i % HEADS;
    int s, d;
    edge_sd(ge, ei, s, d);
    float e = leaky02(asrc[s * HEADS + h] + adst[d * HEADS + h]);
    unsigned u = __float_as_uint(e);
    u = (u & 0x80000000u) ? ~u : (u | 0x80000000u);  // order-preserving f32->u32
    atomicMax(&emax[d * HEADS + h], u);
}

__global__ void edge_sum_kernel(const int* __restrict__ ei,
                                const float* __restrict__ asrc,
                                const float* __restrict__ adst,
                                const unsigned* __restrict__ emax,
                                const float* __restrict__ xp,
                                float* __restrict__ denom,
                                float* __restrict__ agg) {
    const int TOT = (BB * EE + BN) * HEADS;
    int i = blockIdx.x * blockDim.x + threadIdx.x;
    if (i >= TOT) return;
    int ge = i / HEADS, h = i % HEADS;
    int s, d;
    edge_sd(ge, ei, s, d);
    float e = leaky02(asrc[s * HEADS + h] + adst[d * HEADS + h]);
    unsigned u = emax[d * HEADS + h];
    float em = (u & 0x80000000u) ? __uint_as_float(u & 0x7fffffffu)
                                 : __uint_as_float(~u);
    float w = __expf(e - em);
    atomicAdd(&denom[d * HEADS + h], w);
    const float* xps = xp + (size_t)s * GG + h * HD;
    float* ag = agg + ((size_t)d * HEADS + h) * HD;
    for (int c = 0; c < HD; ++c) atomicAdd(&ag[c], w * xps[c]);
}

__global__ void gat_finish_kernel(const float* __restrict__ agg,
                                  const float* __restrict__ denom,
                                  const float* __restrict__ gb,
                                  float* __restrict__ y32,
                                  _Float16* __restrict__ y16) {
    int i = blockIdx.x * blockDim.x + threadIdx.x;
    if (i >= BN * HD) return;
    int n = i / HD, c = i % HD;
    float acc = 0.f;
    for (int h = 0; h < HEADS; ++h)
        acc += agg[((size_t)n * HEADS + h) * HD + c] / (denom[n * HEADS + h] + 1e-16f);
    acc = acc * (1.f / HEADS) + gb[c];
    acc = acc > 0.f ? acc : 0.f;   // relu applied by caller in reference
    y32[i] = acc;
    y16[i] = (_Float16)acc;
}

// ---------------------------------------------------------------------------
// Pool over nodes + final MLP head
// ---------------------------------------------------------------------------
__global__ void pool_kernel(const float* __restrict__ y, float* __restrict__ pooled) {
    int i = blockIdx.x * blockDim.x + threadIdx.x;
    if (i >= BB * HD) return;
    int b = i / HD, c = i % HD;
    float s = 0.f;
    for (int n = 0; n < NN; ++n) s += y[(size_t)(b * NN + n) * HD + c];
    pooled[i] = s / (float)NN;
}

__global__ void head_kernel(const float* __restrict__ pooled,
                            const float* __restrict__ fW1, const float* __restrict__ fb1,
                            const float* __restrict__ fW2, const float* __restrict__ fb2,
                            float* __restrict__ out) {
    __shared__ float hid[BB][HD / 2];
    int tid = threadIdx.x;
    if (tid < BB * (HD / 2)) {
        int b = tid / (HD / 2), j = tid % (HD / 2);
        float s = fb1[j];
        for (int c = 0; c < HD; ++c) s += pooled[b * HD + c] * fW1[j * HD + c];
        hid[b][j] = s > 0.f ? s : 0.f;
    }
    __syncthreads();
    if (tid < BB) {
        float s = fb2[0];
        for (int j = 0; j < HD / 2; ++j) s += hid[tid][j] * fW2[j];
        out[tid] = s;
    }
}

// ---------------------------------------------------------------------------
// Launch
// ---------------------------------------------------------------------------
extern "C" void kernel_launch(void* const* d_in, const int* in_sizes, int n_in,
                              void* d_out, int out_size, void* d_ws, size_t ws_size,
                              hipStream_t stream) {
    (void)in_sizes; (void)n_in; (void)out_size; (void)ws_size;
    const float* x_seq  = (const float*)d_in[0];
    const int*   ei     = (const int*)d_in[1];
    const float* Wih0f  = (const float*)d_in[2];
    const float* Whh0f  = (const float*)d_in[3];
    const float* b0f    = (const float*)d_in[4];
    const float* Wih0b  = (const float*)d_in[5];
    const float* Whh0b  = (const float*)d_in[6];
    const float* b0b    = (const float*)d_in[7];
    const float* Wih1f  = (const float*)d_in[8];
    const float* Whh1f  = (const float*)d_in[9];
    const float* b1f    = (const float*)d_in[10];
    const float* Wih1b  = (const float*)d_in[11];
    const float* Whh1b  = (const float*)d_in[12];
    const float* b1b    = (const float*)d_in[13];
    const float* gW0    = (const float*)d_in[14];
    const float* ga_s0  = (const float*)d_in[15];
    const float* ga_d0  = (const float*)d_in[16];
    const float* gb0    = (const float*)d_in[17];
    const float* gW1    = (const float*)d_in[18];
    const float* ga_s1  = (const float*)d_in[19];
    const float* ga_d1  = (const float*)d_in[20];
    const float* gb1    = (const float*)d_in[21];
    const float* fW1    = (const float*)d_in[22];
    const float* fb1    = (const float*)d_in[23];
    const float* fW2    = (const float*)d_in[24];
    const float* fb2    = (const float*)d_in[25];
    float* out = (float*)d_out;

    // Workspace carve-out (~145 MB total).
    char* ws = (char*)d_ws;
    size_t off = 0;
    auto alloc = [&](size_t bytes) -> char* {
        char* p = ws + off;
        off += (bytes + 255) & ~(size_t)255;
        return p;
    };
    _Float16* h0    = (_Float16*)alloc((size_t)BN * SS * HO * 2);
    _Float16* h1    = (_Float16*)alloc((size_t)BN * SS * HO * 2);
    float*    xp    = (float*)alloc((size_t)BN * GG * 4);
    float*    asrc  = (float*)alloc((size_t)BN * HEADS * 4);
    float*    adst  = (float*)alloc((size_t)BN * HEADS * 4);
    unsigned* emax  = (unsigned*)alloc((size_t)BN * HEADS * 4);
    float*    denom = (float*)alloc((size_t)BN * HEADS * 4);
    float*    agg   = (float*)alloc((size_t)BN * HEADS * HD * 4);
    float*    y32   = (float*)alloc((size_t)BN * HD * 4);
    _Float16* y16   = (_Float16*)alloc((size_t)BN * HD * 2);
    float*    y232  = (float*)alloc((size_t)BN * HD * 4);
    float*    pooled = (float*)alloc((size_t)BB * HD * 4);

    // ---- LSTM layer 0 (input gathered from x_seq), both directions ----
    {
        constexpr int K0 = DD + HD;  // 96
        size_t sm = (size_t)GG * K0 * 2 + 16 * K0 * 2 + 16 * GG * 4 + 16 * HD * 4 + GG * 4;
        lstm_seq_kernel<DD, true><<<BN / 16, 256, sm, stream>>>(x_seq, nullptr, Wih0f, Whh0f, b0f, h0, 0, 0);
        lstm_seq_kernel<DD, true><<<BN / 16, 256, sm, stream>>>(x_seq, nullptr, Wih0b, Whh0b, b0b, h0, HD, 1);
    }
    // ---- LSTM layer 1 (input = h0), both directions ----
    {
        constexpr int K1 = HO + HD;  // 192
        size_t sm = (size_t)GG * K1 * 2 + 16 * K1 * 2 + 16 * GG * 4 + 16 * HD * 4 + GG * 4;
        lstm_seq_kernel<HO, false><<<BN / 16, 256, sm, stream>>>(nullptr, h0, Wih1f, Whh1f, b1f, h1, 0, 0);
        lstm_seq_kernel<HO, false><<<BN / 16, 256, sm, stream>>>(nullptr, h0, Wih1b, Whh1b, b1b, h1, HD, 1);
    }

    const int TOTE = (BB * EE + BN) * HEADS;
    auto run_gat = [&](const _Float16* Ain, long rowStride, bool k128,
                       const float* gW, const float* gas, const float* gad,
                       const float* gb, float* yo32, _Float16* yo16) {
        if (k128)
            gemm_a16_wt_kernel<HO><<<BN / 16, 256, 0, stream>>>(Ain, rowStride, gW, xp);
        else
            gemm_a16_wt_kernel<HD><<<BN / 16, 256, 0, stream>>>(Ain, rowStride, gW, xp);
        hipMemsetAsync(emax, 0, (size_t)BN * HEADS * 4, stream);
        hipMemsetAsync(denom, 0, (size_t)BN * HEADS * 4, stream);
        hipMemsetAsync(agg, 0, (size_t)BN * HEADS * HD * 4, stream);
        alpha_kernel<<<(BN * HEADS + 255) / 256, 256, 0, stream>>>(xp, gas, gad, asrc, adst);
        edge_max_kernel<<<(TOTE + 255) / 256, 256, 0, stream>>>(ei, asrc, adst, emax);
        edge_sum_kernel<<<(TOTE + 255) / 256, 256, 0, stream>>>(ei, asrc, adst, emax, xp, denom, agg);
        gat_finish_kernel<<<(BN * HD + 255) / 256, 256, 0, stream>>>(agg, denom, gb, yo32, yo16);
    };

    // GAT0 consumes last timestep of h1: base pointer offset (S-1)*HO, row stride S*HO.
    run_gat(h1 + (size_t)(SS - 1) * HO, (long)SS * HO, true, gW0, ga_s0, ga_d0, gb0, y32, y16);
    // GAT1 consumes y (f16 copy), row stride HD.
    run_gat(y16, HD, false, gW1, ga_s1, ga_d1, gb1, y232, y16);

    pool_kernel<<<1, 256, 0, stream>>>(y232, pooled);
    head_kernel<<<1, 128, 0, stream>>>(pooled, fW1, fb1, fW2, fb2, out);
}
